// StyleGAN2ResnetGenerator_75153337745849
// MI455X (gfx1250) — compile-verified, matrix operand order checked by reading, not run on hardware
//
#include <hip/hip_runtime.h>
#include <hip/hip_bf16.h>

// ---------------------------------------------------------------------------
// StyleGAN2 resnet generator head for MI455X (gfx1250, wave32).
// bf16 WMMA (v_wmma_f32_16x16x32_bf16) with fp32 accumulate for all matrix work.
// Modulated conv rewritten as:  out = (scale*demod)[b,o] * conv(s[b,i]*x, w)
// Style scales are fused into producer kernels so the conv kernel stages pure
// bytes (async global->LDS) and the WMMA inner loop is
// 2x ds_load_b128 + 2x global_load_b128 + 1 WMMA per K-step.
// ---------------------------------------------------------------------------

typedef __bf16 bf16;
typedef __attribute__((ext_vector_type(16))) __bf16 v16bf;
typedef __attribute__((ext_vector_type(8)))  __bf16 v8bf;
typedef __attribute__((ext_vector_type(8)))  float  v8f;
typedef int v4i __attribute__((vector_size(16)));   // 16B granule for async LDS copy

#define SQRT2F  1.41421356237309515f
#define ISQRT2F 0.70710678118654752f

#if __has_builtin(__builtin_amdgcn_global_load_async_to_lds_b128) && \
    __has_builtin(__builtin_amdgcn_s_wait_asynccnt)
#define HAVE_ASYNC_LDS 1
#else
#define HAVE_ASYNC_LDS 0
#endif

#define AS_GLOBAL __attribute__((address_space(1)))
#define AS_LDS    __attribute__((address_space(3)))

__device__ inline v8bf zero8() {
  v8bf z;
#pragma unroll
  for (int i = 0; i < 8; ++i) z[i] = (bf16)0.f;
  return z;
}

// ---------------------------------------------------------------- cvt fp32->bf16
__global__ void cvt_bf16_k(const float* __restrict__ src, bf16* __restrict__ dst, int n) {
  int i = blockIdx.x * blockDim.x + threadIdx.x;
  if (i < n) dst[i] = (bf16)src[i];
}

// ------------------------------------------------- pack conv weights -> [Cout][Kpad]
// K index = (dy*3+dx)*cin + ci ; optional spatial flip (for transpose-conv path).
__global__ void pack_w_k(const float* __restrict__ w, bf16* __restrict__ dst,
                         int cout, int cin, int kk, int Kpad, int flip) {
  int i = blockIdx.x * blockDim.x + threadIdx.x;
  if (i >= cout * Kpad) return;
  int o = i / Kpad, k = i - o * Kpad;
  int Ktot = kk * kk * cin;
  float v = 0.f;
  if (k < Ktot) {
    int tap = k / cin, ci = k - tap * cin;
    int dy = tap / kk, dx = tap - dy * kk;
    if (flip) { dy = kk - 1 - dy; dx = kk - 1 - dx; }
    v = w[((o * cin + ci) * kk + dy) * kk + dx];
  }
  dst[i] = (bf16)v;
}

// --------------------------------------------- style (equal_linear) + demod factors
__global__ void style_k(const float* __restrict__ g, const float* __restrict__ mod_w,
                        const float* __restrict__ mod_b, const float* __restrict__ w,
                        int cout, int cin, int kk, int demod,
                        float* __restrict__ s_out, float* __restrict__ d_out) {
  int b = blockIdx.x;
  __shared__ float s_sh[64];
  int t = threadIdx.x;                       // 256 threads
  if (t < cin) {
    const float* gb = g + b * 256;
    const float* mw = mod_w + t * 256;
    float acc = 0.f;
    for (int j = 0; j < 256; ++j) acc += gb[j] * mw[j];
    float s = acc * 0.0625f + mod_b[t];
    s_sh[t] = s;
    s_out[b * cin + t] = s;
  }
  __syncthreads();
  float scale = rsqrtf((float)(cin * kk * kk));
  if (t < cout) {
    float dd;
    if (demod) {
      const float* wo = w + t * cin * kk * kk;
      int kkk = kk * kk;
      float ss = 0.f;
      for (int i = 0; i < cin; ++i) {
        float w2 = 0.f;
        for (int k = 0; k < kkk; ++k) { float wv = wo[i * kkk + k]; w2 += wv * wv; }
        ss += w2 * s_sh[i] * s_sh[i];
      }
      dd = scale * rsqrtf(scale * scale * ss + 1e-8f);
    } else {
      dd = scale;
    }
    d_out[b * cout + t] = dd;
  }
}

// ---------------------------------------------------------------- WMMA GEMM (fc)
// A [M][K] bf16 row-major, Bp [N][K] bf16 (pre-transposed), out [M][N] bf16 + bias.
__global__ void gemm_wmma_k(const bf16* __restrict__ A, const bf16* __restrict__ Bp,
                            const float* __restrict__ bias, bf16* __restrict__ out,
                            int M, int N, int K) {
  int wave = threadIdx.x >> 5, lane = threadIdx.x & 31;
  int m0 = blockIdx.x * 16;
  int n  = blockIdx.y * 64 + wave * 16 + (lane & 15);
  int mrow = m0 + (lane & 15);
  int koff = (lane >> 4) * 8;     // 16-bit A/B fragment layout: half-lane K offset
  v8f acc = {};
  for (int kb = 0; kb < K; kb += 32) {
    const bf16* ap = A  + mrow * K + kb + koff;
    const bf16* bp = Bp + n    * K + kb + koff;
    __builtin_prefetch(bp + 64, 0, 0);
    union { v16bf v; v8bf h[2]; } a, bm;
    a.h[0]  = *(const v8bf*)(ap);
    a.h[1]  = *(const v8bf*)(ap + 16);
    bm.h[0] = *(const v8bf*)(bp);
    bm.h[1] = *(const v8bf*)(bp + 16);
    acc = __builtin_amdgcn_wmma_f32_16x16x32_bf16(false, a.v, false, bm.v,
                                                  (short)0, acc, false, false);
  }
#pragma unroll
  for (int r = 0; r < 8; ++r) {
    int m = m0 + r + (lane >> 4) * 8;
    float v = acc[r] + bias[n];
    out[m * N + n] = (bf16)v;
  }
}

// ------------------------------------------- reshape fc2 out [B][512] -> NHWC [B][8][8][8]
__global__ void reshape_x0_k(const bf16* __restrict__ src, bf16* __restrict__ dst) {
  int i = blockIdx.x * blockDim.x + threadIdx.x;     // b*512 + c*64 + h*8 + w
  if (i >= 1024 * 512) return;
  int b = i >> 9, r = i & 511;
  int c = r >> 6, hw = r & 63;
  dst[(b * 64 + hw) * 8 + c] = src[i];
}

// ----------------------------- per-(b,channel) scale copy (style pre-modulation)
__global__ void scale_k(const bf16* __restrict__ in, const float* __restrict__ s,
                        bf16* __restrict__ out, int HW, int C) {
  int b = blockIdx.y;
  int i = blockIdx.x * blockDim.x + threadIdx.x;
  if (i >= HW * C) return;
  int c = i % C;
  size_t idx = (size_t)b * HW * C + i;
  out[idx] = (bf16)((float)in[idx] * s[b * C + c]);
}

// ---------------------------------------------------- generic WMMA 3x3 conv (NHWC)
// Implicit GEMM: M = output pixels (16/tile), N = Cout, K = 9*Cin (zero-padded).
// Input is PRE-SCALED by style. Staging = pure copy (async global->LDS if HW path
// exists). Fragment loads exploit: every aligned 8-element K-run lies inside one
// tap with contiguous channels -> one 16B ds_load_b128 per half fragment.
// Epilogue: *= dd[b,o], +bias, optional lrelu*sqrt2, optional next-layer style.
__global__ void conv_wmma_k(const bf16* __restrict__ in, const bf16* __restrict__ wpk,
                            const float* __restrict__ dd, const float* __restrict__ bias,
                            const float* __restrict__ snext, bf16* __restrict__ out,
                            int Hin, int Win, int Cin, int cinSh,
                            int Hout, int Wout, int Cout, int pad, int Kpad, int act) {
  extern __shared__ bf16 smem[];
  const int b = blockIdx.y;
  const int npix = Hout * Wout;
  const int tile0 = blockIdx.x * 16;
  const int Wp = Win + 2 * pad;
  const int h0 = tile0 / Wout;            // first output row of this tile
  const int rr0 = h0 - pad;               // first staged input row
  const int WpC = Wp * Cin;

#if HAVE_ASYNC_LDS
  // zero pad columns + fully-OOB rows with DS stores (disjoint from async range)
  for (int idx = threadIdx.x; idx < 4 * WpC; idx += blockDim.x) {
    int r = idx / WpC, rem = idx - r * WpC;
    int gr = rr0 + r;
    bool interior = (gr >= 0 && gr < Hin) &&
                    (rem >= pad * Cin) && (rem < (pad + Win) * Cin);
    if (!interior) smem[idx] = (bf16)0.f;
  }
  // async-copy the in-bounds interior rows straight into LDS (16B granules)
  {
    const int rowBytes = Win * Cin * 2;
#pragma unroll
    for (int r = 0; r < 4; ++r) {
      int gr = rr0 + r;
      if (gr < 0 || gr >= Hin) continue;
      const char* gsrc = (const char*)(in + ((size_t)(b * Hin + gr) * Win) * Cin);
      char* ldst = (char*)(smem + r * WpC + pad * Cin);
      for (int ofs = threadIdx.x * 16; ofs < rowBytes; ofs += blockDim.x * 16) {
        __builtin_amdgcn_global_load_async_to_lds_b128(
            (AS_GLOBAL v4i*)(gsrc + ofs), (AS_LDS v4i*)(ldst + ofs), 0, 0);
      }
    }
  }
  __builtin_amdgcn_s_wait_asynccnt(0);
  __syncthreads();
#else
  for (int idx = threadIdx.x; idx < 4 * WpC; idx += blockDim.x) {
    int r = idx / WpC, rem = idx - r * WpC;
    int c = rem / Cin, ci = rem - c * Cin;
    int gr = rr0 + r, gc = c - pad;
    bf16 v = (bf16)0.f;
    if (gr >= 0 && gr < Hin && gc >= 0 && gc < Win)
      v = in[((b * Hin + gr) * Win + gc) * Cin + ci];
    smem[idx] = v;
  }
  __syncthreads();
#endif

  const int wave = threadIdx.x >> 5, lane = threadIdx.x & 31;
  int p = tile0 + (lane & 15);
  if (p >= npix) p = npix - 1;            // keep EXEC all-ones for WMMA
  const int oh = p / Wout, ow = p - oh * Wout;
  const int lrow0 = oh - h0;              // 0..1
  const int koff8 = (lane >> 4) * 8;
  const int n = wave * 16 + (lane & 15);
  const bf16* bp_base = wpk + n * Kpad;
  const int Ktot = 9 * Cin;
  const int cmask = Cin - 1;

  v8f acc = {};
  for (int kb = 0; kb < Kpad; kb += 32) {
    __builtin_prefetch(bp_base + kb + 32, 0, 0);
    union { v16bf v; v8bf h[2]; } a, bm;
#pragma unroll
    for (int half = 0; half < 2; ++half) {
      const int kbase = kb + half * 16 + koff8;
      // 8-run = contiguous channels of a single tap (Cin, koff multiples of 8)
      if (kbase < Ktot) {
        int tap = kbase >> cinSh;
        int ci  = kbase & cmask;
        int dy = tap / 3, dx = tap - dy * 3;
        a.h[half] = *(const v8bf*)&smem[((lrow0 + dy) * Wp + (ow + dx)) * Cin + ci];
      } else {
        a.h[half] = zero8();
      }
      bm.h[half] = *(const v8bf*)(bp_base + kbase);
    }
    acc = __builtin_amdgcn_wmma_f32_16x16x32_bf16(false, a.v, false, bm.v,
                                                  (short)0, acc, false, false);
  }

  const float ds = dd[b * Cout + n];
  const float bv = bias ? bias[n] : 0.f;
  const float sn = snext ? snext[b * Cout + n] : 1.f;
#pragma unroll
  for (int r = 0; r < 8; ++r) {
    int m = tile0 + r + (lane >> 4) * 8;
    if (m < npix) {
      int oh2 = m / Wout, ow2 = m - oh2 * Wout;
      float v = acc[r] * ds + bv;
      if (act) v = (v > 0.f ? v : 0.2f * v) * SQRT2F;
      out[((b * Hout + oh2) * Wout + ow2) * Cout + n] = (bf16)(v * sn);
    }
  }
}

// --------------- zero-insert upsample H -> 2H-1, fused next-layer style scale
__global__ void zup_k(const bf16* __restrict__ in, const float* __restrict__ s,
                      bf16* __restrict__ out, int H, int W, int C) {
  int b = blockIdx.y;
  int Ho = 2 * H - 1, Wo = 2 * W - 1;
  int i = blockIdx.x * blockDim.x + threadIdx.x;
  if (i >= Ho * Wo * C) return;
  int c = i % C; int pp = i / C; int x = pp % Wo; int y = pp / Wo;
  bf16 v = (bf16)0.f;
  if (!(y & 1) && !(x & 1))
    v = (bf16)((float)in[((b * H + (y >> 1)) * W + (x >> 1)) * C + c] * s[b * C + c]);
  out[((b * Ho + y) * Wo + x) * C + c] = v;
}

// -- blur4 (gain 4) depthwise + deferred bias + lrelu*sqrt2 + next-layer style
__global__ void blur_k(const bf16* __restrict__ in, const float* __restrict__ bias,
                       const float* __restrict__ snext, bf16* __restrict__ out,
                       int Hz, int Wz, int C, int act) {
  int b = blockIdx.y;
  int Ho = Hz - 1, Wo = Wz - 1;
  int i = blockIdx.x * blockDim.x + threadIdx.x;
  if (i >= Ho * Wo * C) return;
  int c = i % C; int pp = i / C; int x = pp % Wo; int y = pp / Wo;
  const float wt[4] = {0.25f, 0.75f, 0.75f, 0.25f};   // [1,3,3,1]/4 per dim (gain 4)
  float acc = 0.f;
#pragma unroll
  for (int iy = 0; iy < 4; ++iy) {
    int sy = y - 1 + iy;
    if (sy < 0 || sy >= Hz) continue;
    float ra = 0.f;
#pragma unroll
    for (int ix = 0; ix < 4; ++ix) {
      int sx = x - 1 + ix;
      if (sx < 0 || sx >= Wz) continue;
      ra += wt[ix] * (float)in[((b * Hz + sy) * Wz + sx) * C + c];
    }
    acc += wt[iy] * ra;
  }
  float v = acc + (bias ? bias[c] : 0.f);
  if (act) v = (v > 0.f ? v : 0.2f * v) * SQRT2F;
  if (snext) v *= snext[b * C + c];
  out[((b * Ho + y) * Wo + x) * C + c] = (bf16)v;
}

// -- res-block combine: out = (equal_conv1x1(xin) + res)/sqrt2 (+ scaled copy)
__global__ void skipres_k(const bf16* __restrict__ xin, const float* __restrict__ wsk,
                          const bf16* __restrict__ res, bf16* __restrict__ out,
                          const float* __restrict__ snext, bf16* __restrict__ out_s,
                          int HW, int Cin, int Cout) {
  int b = blockIdx.y;
  int i = blockIdx.x * blockDim.x + threadIdx.x;
  if (i >= HW * Cout) return;
  int o = i % Cout, p = i / Cout;
  float sc = rsqrtf((float)Cin);
  const bf16* xp = xin + (b * HW + p) * Cin;
  float acc = 0.f;
  for (int ci = 0; ci < Cin; ++ci) acc += (float)xp[ci] * wsk[o * Cin + ci];
  size_t oi = (size_t)(b * HW + p) * Cout + o;
  float v = (acc * sc + (float)res[oi]) * ISQRT2F;
  out[oi] = (bf16)v;
  if (out_s) out_s[oi] = (bf16)(v * snext[b * Cout + o]);
}

// ------------- u16 skip pre-path: fused_lrelu(equal_conv1x1(x) + b) at low res
__global__ void skiplo_k(const bf16* __restrict__ xin, const float* __restrict__ wsk,
                         const float* __restrict__ bias, bf16* __restrict__ out,
                         int HW, int Cin, int Cout) {
  int b = blockIdx.y;
  int i = blockIdx.x * blockDim.x + threadIdx.x;
  if (i >= HW * Cout) return;
  int o = i % Cout, p = i / Cout;
  float sc = rsqrtf((float)Cin);
  const bf16* xp = xin + (b * HW + p) * Cin;
  float acc = 0.f;
  for (int ci = 0; ci < Cin; ++ci) acc += (float)xp[ci] * wsk[o * Cin + ci];
  float v = acc * sc + bias[o];
  v = (v > 0.f ? v : 0.2f * v) * SQRT2F;
  out[(b * HW + p) * Cout + o] = (bf16)v;
}

// ------- bilinear-up2(skip_lo) + res, /sqrt2.  skip_lo at HxW, out at 2Hx2W.
__global__ void upcomb_k(const bf16* __restrict__ skiplo, const bf16* __restrict__ res,
                         bf16* __restrict__ out, int H, int W, int C) {
  int b = blockIdx.y;
  int Ho = 2 * H, Wo = 2 * W;
  int i = blockIdx.x * blockDim.x + threadIdx.x;
  if (i >= Ho * Wo * C) return;
  int c = i % C; int pp = i / C; int x = pp % Wo; int y = pp / Wo;
  float fy = 0.5f * y - 0.25f, fx = 0.5f * x - 0.25f;
  int y0 = (int)floorf(fy), x0 = (int)floorf(fx);
  float wy = fy - (float)y0, wx = fx - (float)x0;
  int y0c = y0 < 0 ? 0 : (y0 > H - 1 ? H - 1 : y0);
  int y1t = y0 + 1; int y1c = y1t < 0 ? 0 : (y1t > H - 1 ? H - 1 : y1t);
  int x0c = x0 < 0 ? 0 : (x0 > W - 1 ? W - 1 : x0);
  int x1t = x0 + 1; int x1c = x1t < 0 ? 0 : (x1t > W - 1 ? W - 1 : x1t);
  const bf16* base = skiplo + (size_t)b * H * W * C;
  float g00 = (float)base[(y0c * W + x0c) * C + c];
  float g01 = (float)base[(y0c * W + x1c) * C + c];
  float g10 = (float)base[(y1c * W + x0c) * C + c];
  float g11 = (float)base[(y1c * W + x1c) * C + c];
  float sk = (1.f - wy) * ((1.f - wx) * g00 + wx * g01) +
             wy * ((1.f - wx) * g10 + wx * g11);
  size_t oi = ((size_t)(b * Ho + y) * Wo + x) * C + c;
  float v = (sk + (float)res[oi]) * ISQRT2F;
  out[oi] = (bf16)v;
}

// ---- ToRGB: modulated 1x1 (no demod) -> fp32 NCHW output + bias
__global__ void rgb_k(const bf16* __restrict__ xin, const float* __restrict__ w,
                      const float* __restrict__ s, const float* __restrict__ bias,
                      float* __restrict__ out) {
  int b = blockIdx.y;
  int i = blockIdx.x * blockDim.x + threadIdx.x;   // o*1024 + p
  if (i >= 3 * 1024) return;
  int o = i / 1024, p = i - o * 1024;
  const bf16*  xp = xin + ((size_t)b * 1024 + p) * 32;
  const float* sp = s + b * 32;
  float acc = 0.f;
#pragma unroll
  for (int ci = 0; ci < 32; ++ci) acc += (float)xp[ci] * sp[ci] * w[o * 32 + ci];
  out[((size_t)b * 3 + o) * 1024 + p] = acc * 0.176776695296637f /*1/sqrt(32)*/ + bias[o];
}

// ===========================================================================
extern "C" void kernel_launch(void* const* d_in, const int* in_sizes, int n_in,
                              void* d_out, int out_size, void* d_ws, size_t ws_size,
                              hipStream_t stream) {
  (void)in_sizes; (void)n_in; (void)out_size; (void)ws_size;
  const int B = 1024;

  // ---- inputs (setup_inputs dict order, depth-first) ----
  const float* spatial  = (const float*)d_in[0];
  const float* gcode    = (const float*)d_in[1];
  const float* fc1_w    = (const float*)d_in[2];
  const float* fc1_b    = (const float*)d_in[3];
  const float* fc2_w    = (const float*)d_in[4];
  const float* fc2_b    = (const float*)d_in[5];
  const float* h0c1_w = (const float*)d_in[6];  const float* h0c1_mw = (const float*)d_in[7];
  const float* h0c1_mb = (const float*)d_in[8]; const float* h0c1_b = (const float*)d_in[9];
  const float* h0c2_w = (const float*)d_in[10]; const float* h0c2_mw = (const float*)d_in[11];
  const float* h0c2_mb = (const float*)d_in[12]; const float* h0c2_b = (const float*)d_in[13];
  const float* h0_skw = (const float*)d_in[14];
  const float* h1c1_w = (const float*)d_in[15]; const float* h1c1_mw = (const float*)d_in[16];
  const float* h1c1_mb = (const float*)d_in[17]; const float* h1c1_b = (const float*)d_in[18];
  const float* h1c2_w = (const float*)d_in[19]; const float* h1c2_mw = (const float*)d_in[20];
  const float* h1c2_mb = (const float*)d_in[21]; const float* h1c2_b = (const float*)d_in[22];
  const float* h1_skw = (const float*)d_in[23];
  const float* u8c1_w = (const float*)d_in[24]; const float* u8c1_mw = (const float*)d_in[25];
  const float* u8c1_mb = (const float*)d_in[26]; const float* u8c1_b = (const float*)d_in[27];
  const float* u8c2_w = (const float*)d_in[28]; const float* u8c2_mw = (const float*)d_in[29];
  const float* u8c2_mb = (const float*)d_in[30]; const float* u8c2_b = (const float*)d_in[31];
  const float* u16c1_w = (const float*)d_in[32]; const float* u16c1_mw = (const float*)d_in[33];
  const float* u16c1_mb = (const float*)d_in[34]; const float* u16c1_b = (const float*)d_in[35];
  const float* u16c2_w = (const float*)d_in[36]; const float* u16c2_mw = (const float*)d_in[37];
  const float* u16c2_mb = (const float*)d_in[38]; const float* u16c2_b = (const float*)d_in[39];
  const float* u16_skw = (const float*)d_in[40]; const float* u16_skb = (const float*)d_in[41];
  const float* rgb_w = (const float*)d_in[42];  const float* rgb_mw = (const float*)d_in[43];
  const float* rgb_mb = (const float*)d_in[44]; const float* rgb_bias = (const float*)d_in[45];

  // ---- workspace arena ----
  char* ws = (char*)d_ws;
  size_t off = 0;
  auto alloc = [&](size_t bytes) -> void* {
    void* p = ws + off;
    off = (off + bytes + 255) & ~(size_t)255;
    return p;
  };

  float *s_h0c1 = (float*)alloc(B*8*4),   *d_h0c1 = (float*)alloc(B*32*4);
  float *s_h0c2 = (float*)alloc(B*32*4),  *d_h0c2 = (float*)alloc(B*32*4);
  float *s_h1c1 = (float*)alloc(B*32*4),  *d_h1c1 = (float*)alloc(B*64*4);
  float *s_h1c2 = (float*)alloc(B*64*4),  *d_h1c2 = (float*)alloc(B*64*4);
  float *s_u8c1 = (float*)alloc(B*64*4),  *d_u8c1 = (float*)alloc(B*64*4);
  float *s_u8c2 = (float*)alloc(B*64*4),  *d_u8c2 = (float*)alloc(B*64*4);
  float *s_u16c1 = (float*)alloc(B*64*4), *d_u16c1 = (float*)alloc(B*32*4);
  float *s_u16c2 = (float*)alloc(B*32*4), *d_u16c2 = (float*)alloc(B*32*4);
  float *s_rgb  = (float*)alloc(B*32*4),  *d_rgb  = (float*)alloc(B*3*4);

  bf16* sp_bf   = (bf16*)alloc(B*64*2);
  bf16* fc1w_bf = (bf16*)alloc(512*64*2);
  bf16* fc2w_bf = (bf16*)alloc(512*512*2);
  bf16* wp_h0c1 = (bf16*)alloc(32*96*2);
  bf16* wp_h0c2 = (bf16*)alloc(32*288*2);
  bf16* wp_h1c1 = (bf16*)alloc(64*288*2);
  bf16* wp_h1c2 = (bf16*)alloc(64*576*2);
  bf16* wp_u8c1 = (bf16*)alloc(64*576*2);
  bf16* wp_u8c2 = (bf16*)alloc(64*576*2);
  bf16* wp_u16c1 = (bf16*)alloc(32*576*2);
  bf16* wp_u16c2 = (bf16*)alloc(32*288*2);

  bf16* fc1o = (bf16*)alloc((size_t)B*512*2);
  bf16* fc2o = (bf16*)alloc((size_t)B*512*2);
  bf16* x0   = (bf16*)alloc((size_t)B*512*2);          // [B][8][8][8]
  bf16* xs0  = (bf16*)alloc((size_t)B*512*2);          // style-scaled for h0c1
  bf16* h0m  = (bf16*)alloc((size_t)B*64*32*2);        // scaled for h0c2
  bf16* h0r  = (bf16*)alloc((size_t)B*64*32*2);
  bf16* h0o  = (bf16*)alloc((size_t)B*64*32*2);
  bf16* h0os = (bf16*)alloc((size_t)B*64*32*2);        // scaled for h1c1
  bf16* h1m  = (bf16*)alloc((size_t)B*64*64*2);        // scaled for h1c2
  bf16* h1r  = (bf16*)alloc((size_t)B*64*64*2);
  bf16* h1o  = (bf16*)alloc((size_t)B*64*64*2);
  bf16* sk16 = (bf16*)alloc((size_t)B*256*32*2);       // u16 skip @16x16

  bf16* bufA = (bf16*)alloc((size_t)B*31*31*64*2);     // z15 / mid16 / z31 / mid32
  bf16* bufB = (bf16*)alloc((size_t)B*32*32*64*2);     // c17 / u8o / blur32
  bf16* bufC = (bf16*)alloc((size_t)B*33*33*32*2);     // blur16 / c33 / u16o

  auto cdiv = [](int a, int b) { return (a + b - 1) / b; };

  // ---- 0) conversions / packing ----
  { int n = B*64;      cvt_bf16_k<<<cdiv(n,256),256,0,stream>>>(spatial, sp_bf, n); }
  { int n = 512*64;    cvt_bf16_k<<<cdiv(n,256),256,0,stream>>>(fc1_w, fc1w_bf, n); }
  { int n = 512*512;   cvt_bf16_k<<<cdiv(n,256),256,0,stream>>>(fc2_w, fc2w_bf, n); }
  auto pack = [&](const float* w, bf16* dst, int cout, int cin, int Kpad, int flip) {
    int n = cout * Kpad;
    pack_w_k<<<cdiv(n,256),256,0,stream>>>(w, dst, cout, cin, 3, Kpad, flip);
  };
  pack(h0c1_w, wp_h0c1, 32, 8, 96, 0);
  pack(h0c2_w, wp_h0c2, 32, 32, 288, 0);
  pack(h1c1_w, wp_h1c1, 64, 32, 288, 0);
  pack(h1c2_w, wp_h1c2, 64, 64, 576, 0);
  pack(u8c1_w, wp_u8c1, 64, 64, 576, 1);     // pre-flipped (transpose-conv path)
  pack(u8c2_w, wp_u8c2, 64, 64, 576, 0);
  pack(u16c1_w, wp_u16c1, 32, 64, 576, 1);   // pre-flipped
  pack(u16c2_w, wp_u16c2, 32, 32, 288, 0);

  // ---- 1) styles + demod factors ----
  auto style = [&](const float* mw, const float* mb, const float* w,
                   int cout, int cin, int kk, int demod, float* s, float* d) {
    style_k<<<B,256,0,stream>>>(gcode, mw, mb, w, cout, cin, kk, demod, s, d);
  };
  style(h0c1_mw, h0c1_mb, h0c1_w, 32, 8, 3, 1, s_h0c1, d_h0c1);
  style(h0c2_mw, h0c2_mb, h0c2_w, 32, 32, 3, 1, s_h0c2, d_h0c2);
  style(h1c1_mw, h1c1_mb, h1c1_w, 64, 32, 3, 1, s_h1c1, d_h1c1);
  style(h1c2_mw, h1c2_mb, h1c2_w, 64, 64, 3, 1, s_h1c2, d_h1c2);
  style(u8c1_mw, u8c1_mb, u8c1_w, 64, 64, 3, 1, s_u8c1, d_u8c1);
  style(u8c2_mw, u8c2_mb, u8c2_w, 64, 64, 3, 1, s_u8c2, d_u8c2);
  style(u16c1_mw, u16c1_mb, u16c1_w, 32, 64, 3, 1, s_u16c1, d_u16c1);
  style(u16c2_mw, u16c2_mb, u16c2_w, 32, 32, 3, 1, s_u16c2, d_u16c2);
  style(rgb_mw, rgb_mb, rgb_w, 3, 32, 1, 0, s_rgb, d_rgb);

  // ---- 2) FC path: spatial @ fc1 -> @ fc2 -> reshape NHWC + style pre-scale ----
  gemm_wmma_k<<<dim3(64,8),128,0,stream>>>(sp_bf, fc1w_bf, fc1_b, fc1o, B, 512, 64);
  gemm_wmma_k<<<dim3(64,8),128,0,stream>>>(fc1o, fc2w_bf, fc2_b, fc2o, B, 512, 512);
  reshape_x0_k<<<cdiv(B*512,256),256,0,stream>>>(fc2o, x0);
  scale_k<<<dim3(cdiv(64*8,256),B),256,0,stream>>>(x0, s_h0c1, xs0, 64, 8);

  auto conv = [&](const bf16* in, const bf16* wpk, const float* d, const float* bias,
                  const float* snext, bf16* out, int Hin, int Win, int Cin, int cinSh,
                  int pad, int Cout, int Kpad, int act) {
    int Hout = Hin + 2*pad - 2, Wout = Win + 2*pad - 2;
    int tiles = cdiv(Hout*Wout, 16);
    size_t lds = (size_t)4 * (Win + 2*pad) * Cin * sizeof(bf16);
    conv_wmma_k<<<dim3(tiles,B), 32*(Cout/16), lds, stream>>>(
        in, wpk, d, bias, snext, out, Hin, Win, Cin, cinSh, Hout, Wout, Cout, pad, Kpad, act);
  };

  // ---- 3) h0: 8 -> 32 ch @ 8x8 ----
  conv(xs0, wp_h0c1, d_h0c1, h0c1_b, s_h0c2, h0m, 8,8, 8,3, 1, 32, 96, 1);
  conv(h0m, wp_h0c2, d_h0c2, h0c2_b, nullptr, h0r, 8,8, 32,5, 1, 32, 288, 1);
  skipres_k<<<dim3(cdiv(64*32,256),B),256,0,stream>>>(x0, h0_skw, h0r, h0o,
                                                      s_h1c1, h0os, 64, 8, 32);

  // ---- 4) h1: 32 -> 64 ch @ 8x8 ----
  conv(h0os, wp_h1c1, d_h1c1, h1c1_b, s_h1c2, h1m, 8,8, 32,5, 1, 64, 288, 1);
  conv(h1m,  wp_h1c2, d_h1c2, h1c2_b, nullptr, h1r, 8,8, 64,6, 1, 64, 576, 1);
  skipres_k<<<dim3(cdiv(64*64,256),B),256,0,stream>>>(h0o, h1_skw, h1r, h1o,
                                                      nullptr, nullptr, 64, 32, 64);

  // ---- 5) u8: 64 -> 64 ch, 8x8 -> 16x16 ----
  bf16* z15    = bufA;   // [B][15][15][64], pre-scaled by s_u8c1
  bf16* c17    = bufB;   // [B][17][17][64]
  bf16* blur16 = bufC;   // [B][16][16][64], pre-scaled by s_u8c2
  bf16* mid16  = bufA;   // z15 dead
  bf16* u8o    = bufB;   // c17 dead
  zup_k<<<dim3(cdiv(15*15*64,256),B),256,0,stream>>>(h1o, s_u8c1, z15, 8, 8, 64);
  conv(z15, wp_u8c1, d_u8c1, nullptr, nullptr, c17, 15,15, 64,6, 2, 64, 576, 0);
  blur_k<<<dim3(cdiv(16*16*64,256),B),256,0,stream>>>(c17, u8c1_b, s_u8c2, blur16,
                                                      17, 17, 64, 1);
  conv(blur16, wp_u8c2, d_u8c2, u8c2_b, nullptr, mid16, 16,16, 64,6, 1, 64, 576, 1);
  upcomb_k<<<dim3(cdiv(16*16*64,256),B),256,0,stream>>>(h1o, mid16, u8o, 8, 8, 64);

  // ---- 6) u16: 64 -> 32 ch, 16x16 -> 32x32 ----
  skiplo_k<<<dim3(cdiv(256*32,256),B),256,0,stream>>>(u8o, u16_skw, u16_skb, sk16,
                                                      256, 64, 32);
  bf16* z31    = bufA;   // mid16 dead; pre-scaled by s_u16c1
  bf16* c33    = bufC;   // blur16 dead
  bf16* blur32 = bufB;   // u8o dead after z31 + sk16; pre-scaled by s_u16c2
  bf16* mid32  = bufA;   // z31 dead
  bf16* u16o   = bufC;   // c33 dead
  zup_k<<<dim3(cdiv(31*31*64,256),B),256,0,stream>>>(u8o, s_u16c1, z31, 16, 16, 64);
  conv(z31, wp_u16c1, d_u16c1, nullptr, nullptr, c33, 31,31, 64,6, 2, 32, 576, 0);
  blur_k<<<dim3(cdiv(32*32*32,256),B),256,0,stream>>>(c33, u16c1_b, s_u16c2, blur32,
                                                      33, 33, 32, 1);
  conv(blur32, wp_u16c2, d_u16c2, u16c2_b, nullptr, mid32, 32,32, 32,5, 1, 32, 288, 1);
  upcomb_k<<<dim3(cdiv(32*32*32,256),B),256,0,stream>>>(sk16, mid32, u16o, 16, 16, 32);

  // ---- 7) ToRGB (no demod) -> fp32 NCHW output ----
  rgb_k<<<dim3(cdiv(3*1024,256),B),256,0,stream>>>(u16o, rgb_w, s_rgb, rgb_bias,
                                                   (float*)d_out);
}
